// WiKG_49538152792237
// MI455X (gfx1250) — matrix operand order repaired and checked by reading, not compiled, not run on hardware
//
#include <hip/hip_runtime.h>
#include <hip/hip_bf16.h>

typedef __attribute__((ext_vector_type(16))) __bf16 v16bf;
typedef __attribute__((ext_vector_type(8)))  float  v8f;
typedef __attribute__((ext_vector_type(4)))  float  v4f;

#define NN   8192
#define DIN  384
#define DH   512
#define TOPK 6

static __device__ __forceinline__ v8f wmma_bf16(v16bf a, v16bf b, v8f c) {
    return __builtin_amdgcn_wmma_f32_16x16x32_bf16(false, a, false, b, (short)0, c, false, false);
}
static __device__ __forceinline__ float lrelu(float x) { return x > 0.0f ? x : 0.01f * x; }
static __device__ __forceinline__ v8f v8f_zero() {
    v8f z;
    #pragma unroll
    for (int i = 0; i < 8; ++i) z[i] = 0.0f;
    return z;
}

// ---------------------------------------------------------------------------
// Generic GEMM: C[M,N] = epilogue(A[M,K] @ W[K,N] + bias)
// bf16 split (hi+lo) -> 3 WMMAs per tile, f32 accumulate (~fp32 accuracy).
// mode 0: +bias; mode 1: lrelu(+bias); mode 2: C += lrelu(+bias)
// Block: 256 thr (8 waves). Tile 128(M) x 64(N), K-step 32. float4 staging.
// ---------------------------------------------------------------------------
__global__ __launch_bounds__(256)
void gemm_split_kernel(const float* __restrict__ A, const float* __restrict__ W,
                       const float* __restrict__ bias, float* __restrict__ C,
                       int M, int Kd, int Nd, int mode)
{
    __shared__ __bf16 Ah[128][34];
    __shared__ __bf16 Al[128][34];
    __shared__ __bf16 Bh[64][34];
    __shared__ __bf16 Bl[64][34];

    const int tid  = threadIdx.x;
    const int wave = tid >> 5;
    const int lane = tid & 31;
    const int ln   = lane & 15;
    const int hs   = lane >> 4;
    const int m0   = blockIdx.y * 128;
    const int n0   = blockIdx.x * 64;

    v8f acc[4];
    #pragma unroll
    for (int nt = 0; nt < 4; ++nt) acc[nt] = v8f_zero();

    for (int k0 = 0; k0 < Kd; k0 += 32) {
        __syncthreads();
        // Stage A tile [128 x 32] as hi/lo bf16 (16B global loads)
        for (int t = tid; t < 1024; t += 256) {
            int r = t >> 3, c4 = (t & 7) * 4;
            v4f xv = *(const v4f*)&A[(size_t)(m0 + r) * Kd + k0 + c4];
            #pragma unroll
            for (int ii = 0; ii < 4; ++ii) {
                float f = xv[ii];
                __bf16 hh = (__bf16)f;
                Ah[r][c4 + ii] = hh;
                Al[r][c4 + ii] = (__bf16)(f - (float)hh);
            }
        }
        // Stage W tile [32 x 64], transposed in LDS as [n][k]
        for (int t = tid; t < 512; t += 256) {
            int kk = t >> 4, n4 = (t & 15) * 4;
            v4f xv = *(const v4f*)&W[(size_t)(k0 + kk) * Nd + n0 + n4];
            #pragma unroll
            for (int ii = 0; ii < 4; ++ii) {
                float f = xv[ii];
                __bf16 hh = (__bf16)f;
                Bh[n4 + ii][kk] = hh;
                Bl[n4 + ii][kk] = (__bf16)(f - (float)hh);
            }
        }
        __syncthreads();

        // A fragment (rows wave*16..+15): lane ln = row; elem i -> K = 16*hs + i (contiguous, B uses the same)
        v16bf a_hi, a_lo;
        #pragma unroll
        for (int i = 0; i < 16; ++i) {
            int kk = i + hs * 16;
            a_hi[i] = Ah[wave * 16 + ln][kk];
            a_lo[i] = Al[wave * 16 + ln][kk];
        }
        #pragma unroll
        for (int nt = 0; nt < 4; ++nt) {
            v16bf b_hi, b_lo;
            #pragma unroll
            for (int i = 0; i < 16; ++i) {
                int kk = i + hs * 16;
                b_hi[i] = Bh[nt * 16 + ln][kk];
                b_lo[i] = Bl[nt * 16 + ln][kk];
            }
            acc[nt] = wmma_bf16(a_hi, b_hi, acc[nt]);
            acc[nt] = wmma_bf16(a_hi, b_lo, acc[nt]);
            acc[nt] = wmma_bf16(a_lo, b_hi, acc[nt]);
        }
    }

    // Epilogue: D elem i of lane -> row = i + 8*hs, col = ln (within 16x16 tile)
    #pragma unroll
    for (int nt = 0; nt < 4; ++nt) {
        #pragma unroll
        for (int i = 0; i < 8; ++i) {
            int row = m0 + wave * 16 + hs * 8 + i;
            int col = n0 + nt * 16 + ln;
            float v = acc[nt][i] + bias[col];
            if (mode >= 1) v = lrelu(v);
            float* p = &C[(size_t)row * Nd + col];
            if (mode == 2) v += *p;
            *p = v;
        }
    }
}

// ---------------------------------------------------------------------------
// Fused attention scores + top-6:  score[i][j] = (e_h[i]*scale) . e_t[j]
// 64 rows / block (halves e_t re-stream traffic), 128-column panels,
// parallel 4-thread/row scan with exact tie-aware 24->6 merge.
// Never materializes the NxN score matrix.
// ---------------------------------------------------------------------------
__global__ __launch_bounds__(256)
void attn_topk_kernel(const float* __restrict__ eh, const float* __restrict__ et,
                      float* __restrict__ tkw, int* __restrict__ tki)
{
    __shared__ __bf16 Ah[64][34];
    __shared__ __bf16 Al[64][34];
    __shared__ __bf16 Bh[128][34];
    __shared__ __bf16 Bl[128][34];
    __shared__ float  Ss[64][132];   // scores; tail reused as merge scratch

    const int tid  = threadIdx.x;
    const int wave = tid >> 5;
    const int lane = tid & 31;
    const int ln   = lane & 15;
    const int hs   = lane >> 4;
    const int row0 = blockIdx.x * 64;
    const int rt   = wave & 3;        // row-tile 0..3
    const int cg   = wave >> 2;       // col-tile group: tiles cg*4 .. cg*4+3
    const int srow = tid & 63;        // scan row
    const int sq   = tid >> 6;        // scan quarter (cols sq*32 .. +32)
    const float scale = rsqrtf((float)DH);

    float tw[TOPK]; int tix[TOPK];
    #pragma unroll
    for (int k = 0; k < TOPK; ++k) { tw[k] = -3.4e38f; tix[k] = 0; }

    for (int cb = 0; cb < NN; cb += 128) {
        v8f acc[4];
        #pragma unroll
        for (int j = 0; j < 4; ++j) acc[j] = v8f_zero();

        for (int k0 = 0; k0 < DH; k0 += 32) {
            __syncthreads();
            // A chunk [64 x 32] (scale folded into split), 16B loads
            for (int t = tid; t < 512; t += 256) {
                int r = t >> 3, c4 = (t & 7) * 4;
                v4f xv = *(const v4f*)&eh[(size_t)(row0 + r) * DH + k0 + c4];
                #pragma unroll
                for (int ii = 0; ii < 4; ++ii) {
                    float f = xv[ii] * scale;
                    __bf16 hh = (__bf16)f;
                    Ah[r][c4 + ii] = hh;
                    Al[r][c4 + ii] = (__bf16)(f - (float)hh);
                }
            }
            // B chunk [128 x 32]: e_t rows cb..cb+127 ARE the transposed columns
            for (int t = tid; t < 1024; t += 256) {
                int r = t >> 3, c4 = (t & 7) * 4;
                v4f xv = *(const v4f*)&et[(size_t)(cb + r) * DH + k0 + c4];
                #pragma unroll
                for (int ii = 0; ii < 4; ++ii) {
                    float f = xv[ii];
                    __bf16 hh = (__bf16)f;
                    Bh[r][c4 + ii] = hh;
                    Bl[r][c4 + ii] = (__bf16)(f - (float)hh);
                }
            }
            __syncthreads();

            v16bf a_hi, a_lo;
            #pragma unroll
            for (int i = 0; i < 16; ++i) {
                int kk = i + hs * 16;
                a_hi[i] = Ah[rt * 16 + ln][kk];
                a_lo[i] = Al[rt * 16 + ln][kk];
            }
            #pragma unroll
            for (int j = 0; j < 4; ++j) {
                v16bf b_hi, b_lo;
                #pragma unroll
                for (int i = 0; i < 16; ++i) {
                    int kk = i + hs * 16;
                    b_hi[i] = Bh[(cg * 4 + j) * 16 + ln][kk];
                    b_lo[i] = Bl[(cg * 4 + j) * 16 + ln][kk];
                }
                acc[j] = wmma_bf16(a_hi, b_hi, acc[j]);
                acc[j] = wmma_bf16(a_hi, b_lo, acc[j]);
                acc[j] = wmma_bf16(a_lo, b_hi, acc[j]);
            }
        }
        __syncthreads();   // prior scan of Ss finished
        #pragma unroll
        for (int j = 0; j < 4; ++j) {
            #pragma unroll
            for (int i = 0; i < 8; ++i) {
                Ss[rt * 16 + hs * 8 + i][(cg * 4 + j) * 16 + ln] = acc[j][i];
            }
        }
        __syncthreads();
        // parallel scan: 4 threads per row, 32 columns each, private top-6
        for (int c = sq * 32; c < sq * 32 + 32; ++c) {
            float s = Ss[srow][c];
            if (s > tw[TOPK - 1]) {
                tw[TOPK - 1] = s; tix[TOPK - 1] = cb + c;
                #pragma unroll
                for (int p = TOPK - 1; p > 0; --p) {
                    if (tw[p] > tw[p - 1]) {
                        float tf = tw[p]; tw[p] = tw[p - 1]; tw[p - 1] = tf;
                        int   tj = tix[p]; tix[p] = tix[p - 1]; tix[p - 1] = tj;
                    }
                }
            }
        }
    }

    // stash 4 partial top-6 lists per row into Ss, then exact merge
    __syncthreads();
    #pragma unroll
    for (int k = 0; k < TOPK; ++k) {
        Ss[srow][sq * TOPK + k] = tw[k];
        ((int*)&Ss[srow][32])[sq * TOPK + k] = tix[k];
    }
    __syncthreads();
    if (tid < 64) {
        float cs[4 * TOPK]; int ci[4 * TOPK];
        #pragma unroll
        for (int j = 0; j < 4 * TOPK; ++j) {
            cs[j] = Ss[tid][j];
            ci[j] = ((int*)&Ss[tid][32])[j];
        }
        // selection of top-6: score desc, tie -> lower index (lax.top_k order)
        for (int s = 0; s < TOPK; ++s) {
            int best = 0;
            float bs = -3.4e38f; int bi = 0x7fffffff;
            for (int j = 0; j < 4 * TOPK; ++j) {
                if (ci[j] < 0) continue;
                if (cs[j] > bs || (cs[j] == bs && ci[j] < bi)) { bs = cs[j]; bi = ci[j]; best = j; }
            }
            tkw[(size_t)(row0 + tid) * TOPK + s] = bs;
            tki[(size_t)(row0 + tid) * TOPK + s] = bi;
            ci[best] = -1;   // mark consumed
        }
    }
}

// ---- column mean of x over nodes ----
__global__ __launch_bounds__(256)
void colmean_kernel(const float* __restrict__ x, float* __restrict__ mean)
{
    int c = blockIdx.x * 256 + threadIdx.x;
    if (c >= DH) return;
    float s = 0.0f;
    for (int n = 0; n < NN; ++n) s += x[(size_t)n * DH + c];
    mean[c] = s * (1.0f / (float)NN);
}

// ---- x = (x + mean)*0.5 ----
__global__ __launch_bounds__(256)
void mixmean_kernel(float* __restrict__ x, const float* __restrict__ mean)
{
    size_t i = (size_t)blockIdx.x * 256 + threadIdx.x;
    if (i >= (size_t)NN * DH) return;
    x[i] = (x[i] + mean[i % DH]) * 0.5f;
}

// ---- neighbor aggregation: softmax(topk), tanh gate, e_Nh; emits u,v ----
__global__ __launch_bounds__(256)
void aggregate_kernel(const float* __restrict__ eh, const float* __restrict__ et,
                      const float* __restrict__ tkw, const int* __restrict__ tki,
                      float* __restrict__ u, float* __restrict__ v)
{
    const int wave = threadIdx.x >> 5;
    const int lane = threadIdx.x & 31;
    const int row  = blockIdx.x * 8 + wave;

    float w[TOPK], p[TOPK]; int id[TOPK];
    #pragma unroll
    for (int k = 0; k < TOPK; ++k) {
        w[k]  = tkw[(size_t)row * TOPK + k];
        id[k] = tki[(size_t)row * TOPK + k];
    }
    float mx = w[0];
    #pragma unroll
    for (int k = 1; k < TOPK; ++k) mx = fmaxf(mx, w[k]);
    float sum = 0.0f;
    #pragma unroll
    for (int k = 0; k < TOPK; ++k) { p[k] = expf(w[k] - mx); sum += p[k]; }
    float inv = 1.0f / sum;
    #pragma unroll
    for (int k = 0; k < TOPK; ++k) p[k] *= inv;

    float ka[TOPK];
    #pragma unroll
    for (int k = 0; k < TOPK; ++k) ka[k] = 0.0f;
    const float* ehrow = eh + (size_t)row * DH;
    for (int d = lane; d < DH; d += 32) {
        float e = ehrow[d];
        #pragma unroll
        for (int k = 0; k < TOPK; ++k) {
            float nb  = et[(size_t)id[k] * DH + d];
            float ehr = p[k] * nb + (1.0f - p[k]) * e;
            ka[k] += nb * tanhf(e + ehr);
        }
    }
    #pragma unroll
    for (int k = 0; k < TOPK; ++k)
        for (int off = 16; off > 0; off >>= 1) ka[k] += __shfl_xor(ka[k], off, 32);

    float kmx = ka[0];
    #pragma unroll
    for (int k = 1; k < TOPK; ++k) kmx = fmaxf(kmx, ka[k]);
    float ksum = 0.0f, kp[TOPK];
    #pragma unroll
    for (int k = 0; k < TOPK; ++k) { kp[k] = expf(ka[k] - kmx); ksum += kp[k]; }
    float kinv = 1.0f / ksum;
    #pragma unroll
    for (int k = 0; k < TOPK; ++k) kp[k] *= kinv;

    for (int d = lane; d < DH; d += 32) {
        float e = ehrow[d];
        float enh = 0.0f;
        #pragma unroll
        for (int k = 0; k < TOPK; ++k) enh += kp[k] * et[(size_t)id[k] * DH + d];
        u[(size_t)row * DH + d] = e + enh;
        v[(size_t)row * DH + d] = e * enh;
    }
}

// ---- a[row] = s[row,:256] . att2_w + att2_b  (s already lrelu'd) ----
__global__ __launch_bounds__(256)
void att2_kernel(const float* __restrict__ s, const float* __restrict__ w2,
                 const float* __restrict__ b2, float* __restrict__ a)
{
    const int wave = threadIdx.x >> 5;
    const int lane = threadIdx.x & 31;
    const int row  = blockIdx.x * 8 + wave;
    float acc = 0.0f;
    for (int c = lane; c < DH / 2; c += 32) acc += s[(size_t)row * (DH / 2) + c] * w2[c];
    for (int off = 16; off > 0; off >>= 1) acc += __shfl_xor(acc, off, 32);
    if (lane == 0) a[row] = acc + b2[0];
}

// ---- softmax over 8192 nodes -> g ----
__global__ __launch_bounds__(256)
void softmax_nodes_kernel(const float* __restrict__ a, float* __restrict__ g)
{
    __shared__ float red[256];
    int tid = threadIdx.x;
    float mx = -3.4e38f;
    for (int n = tid; n < NN; n += 256) mx = fmaxf(mx, a[n]);
    red[tid] = mx; __syncthreads();
    for (int s = 128; s > 0; s >>= 1) { if (tid < s) red[tid] = fmaxf(red[tid], red[tid + s]); __syncthreads(); }
    mx = red[0]; __syncthreads();
    float sum = 0.0f;
    for (int n = tid; n < NN; n += 256) sum += expf(a[n] - mx);
    red[tid] = sum; __syncthreads();
    for (int s = 128; s > 0; s >>= 1) { if (tid < s) red[tid] += red[tid + s]; __syncthreads(); }
    float inv = 1.0f / red[0];
    for (int n = tid; n < NN; n += 256) g[n] = expf(a[n] - mx) * inv;
}

// ---- h[c] = sum_n g[n]*emb[n][c] ----
__global__ __launch_bounds__(256)
void readout_kernel(const float* __restrict__ g, const float* __restrict__ emb,
                    float* __restrict__ h)
{
    int c = blockIdx.x * 256 + threadIdx.x;
    float acc = 0.0f;
    for (int n = 0; n < NN; ++n) acc += g[n] * emb[(size_t)n * DH + c];
    h[c] = acc;
}

// ---- layernorm + final fc -> logits[2] ----
__global__ __launch_bounds__(256)
void final_kernel(const float* __restrict__ h, const float* __restrict__ ln_g,
                  const float* __restrict__ ln_b, const float* __restrict__ fc_w,
                  const float* __restrict__ fc_b, float* __restrict__ out)
{
    __shared__ float red[256];
    int tid = threadIdx.x;
    float v0 = h[tid], v1 = h[tid + 256];
    red[tid] = v0 + v1; __syncthreads();
    for (int s = 128; s > 0; s >>= 1) { if (tid < s) red[tid] += red[tid + s]; __syncthreads(); }
    float mu = red[0] * (1.0f / (float)DH);
    __syncthreads();
    float d0 = v0 - mu, d1 = v1 - mu;
    red[tid] = d0 * d0 + d1 * d1; __syncthreads();
    for (int s = 128; s > 0; s >>= 1) { if (tid < s) red[tid] += red[tid + s]; __syncthreads(); }
    float var = red[0] * (1.0f / (float)DH);
    __syncthreads();
    float rstd = rsqrtf(var + 1e-5f);
    float hn0 = d0 * rstd * ln_g[tid] + ln_b[tid];
    float hn1 = d1 * rstd * ln_g[tid + 256] + ln_b[tid + 256];
    for (int j = 0; j < 2; ++j) {
        red[tid] = hn0 * fc_w[tid * 2 + j] + hn1 * fc_w[(tid + 256) * 2 + j];
        __syncthreads();
        for (int s = 128; s > 0; s >>= 1) { if (tid < s) red[tid] += red[tid + s]; __syncthreads(); }
        if (tid == 0) out[j] = red[0] + fc_b[j];
        __syncthreads();
    }
}

extern "C" void kernel_launch(void* const* d_in, const int* in_sizes, int n_in,
                              void* d_out, int out_size, void* d_ws, size_t ws_size,
                              hipStream_t stream)
{
    const float* x_s    = (const float*)d_in[0];
    const float* fc1_w  = (const float*)d_in[1];
    const float* fc1_b  = (const float*)d_in[2];
    const float* wh_w   = (const float*)d_in[3];
    const float* wh_b   = (const float*)d_in[4];
    const float* wt_w   = (const float*)d_in[5];
    const float* wt_b   = (const float*)d_in[6];
    const float* l1_w   = (const float*)d_in[7];
    const float* l1_b   = (const float*)d_in[8];
    const float* l2_w   = (const float*)d_in[9];
    const float* l2_b   = (const float*)d_in[10];
    const float* att1_w = (const float*)d_in[11];
    const float* att1_b = (const float*)d_in[12];
    const float* att2_w = (const float*)d_in[13];
    const float* att2_b = (const float*)d_in[14];
    const float* ln_g   = (const float*)d_in[15];
    const float* ln_b   = (const float*)d_in[16];
    const float* fc_w   = (const float*)d_in[17];
    const float* fc_b   = (const float*)d_in[18];
    float* out = (float*)d_out;

    float* wsf = (float*)d_ws;
    const size_t NDH = (size_t)NN * DH;
    float* xbuf  = wsf;              // x, later u = e_h + e_Nh
    float* ehb   = wsf + NDH;        // e_h, later emb
    float* etb   = wsf + 2 * NDH;    // e_t, later s (att1 out)
    float* vbuf  = wsf + 3 * NDH;    // v = e_h * e_Nh
    float* meanb = wsf + 4 * NDH;
    float* tkw   = meanb + DH;
    int*   tki   = (int*)(tkw + (size_t)NN * TOPK);
    float* abuf  = (float*)(tki + (size_t)NN * TOPK);
    float* gbuf  = abuf + NN;
    float* hbuf  = gbuf + NN;

    // 1) x = lrelu(x_s @ fc1_w + b)
    gemm_split_kernel<<<dim3(DH / 64, NN / 128), 256, 0, stream>>>(x_s, fc1_w, fc1_b, xbuf, NN, DIN, DH, 1);
    // 2) x = (x + mean(x))*0.5
    colmean_kernel<<<(DH + 255) / 256, 256, 0, stream>>>(xbuf, meanb);
    mixmean_kernel<<<(unsigned)((NDH + 255) / 256), 256, 0, stream>>>(xbuf, meanb);
    // 3) e_h, e_t
    gemm_split_kernel<<<dim3(DH / 64, NN / 128), 256, 0, stream>>>(xbuf, wh_w, wh_b, ehb, NN, DH, DH, 0);
    gemm_split_kernel<<<dim3(DH / 64, NN / 128), 256, 0, stream>>>(xbuf, wt_w, wt_b, etb, NN, DH, DH, 0);
    // 4) fused scores + top-6 (64 rows per block)
    attn_topk_kernel<<<NN / 64, 256, 0, stream>>>(ehb, etb, tkw, tki);
    // 5) neighbor aggregation -> u (reuse xbuf), v
    aggregate_kernel<<<NN / 8, 256, 0, stream>>>(ehb, etb, tkw, tki, xbuf, vbuf);
    // 6) emb = lrelu(u@l1+b) + lrelu(v@l2+b)   (emb reuses ehb)
    gemm_split_kernel<<<dim3(DH / 64, NN / 128), 256, 0, stream>>>(xbuf, l1_w, l1_b, ehb, NN, DH, DH, 1);
    gemm_split_kernel<<<dim3(DH / 64, NN / 128), 256, 0, stream>>>(vbuf, l2_w, l2_b, ehb, NN, DH, DH, 2);
    // 7) s = lrelu(emb@att1+b)  (reuses etb); a = s@att2 + b
    gemm_split_kernel<<<dim3((DH / 2) / 64, NN / 128), 256, 0, stream>>>(ehb, att1_w, att1_b, etb, NN, DH, DH / 2, 1);
    att2_kernel<<<NN / 8, 256, 0, stream>>>(etb, att2_w, att2_b, abuf);
    // 8) g = softmax(a); h = sum g*emb
    softmax_nodes_kernel<<<1, 256, 0, stream>>>(abuf, gbuf);
    readout_kernel<<<DH / 256, 256, 0, stream>>>(gbuf, ehb, hbuf);
    // 9) layernorm + fc
    final_kernel<<<1, 256, 0, stream>>>(hbuf, ln_g, ln_b, fc_w, fc_b, out);
}
